// MultiheadSelfAttention_1202590843145
// MI455X (gfx1250) — matrix operand order repaired and checked by reading, not compiled, not run on hardware
//
#include <hip/hip_runtime.h>

// ---------------------------------------------------------------------------
// Multi-head self-attention over ragged scenes, f16 WMMA pipeline for gfx1250.
// B=2048 scenes, A=64 agents, D=128, H=4 heads, HD=32.
// ---------------------------------------------------------------------------

typedef __attribute__((ext_vector_type(16))) _Float16 v16h;
typedef __attribute__((ext_vector_type(8)))  _Float16 v8h;
typedef __attribute__((ext_vector_type(4)))  _Float16 v4h;
typedef __attribute__((ext_vector_type(8)))  float    v8f;
typedef __attribute__((ext_vector_type(4)))  float    v4f;

#define NSCENE 2048
#define AMAX   64
#define DIM    128
#define HEADS  4
#define HD     32
#define XS     136   // row stride (halves) for 128-wide f16 tiles (pad 8 -> 272B, shifts banks)
#define AS     72    // row stride (halves) for 64-wide f16 tiles (attn, V^T)
#define SFS    68    // row stride (floats) for f32 score scratch

union FragU { v16h v; v8h h[2]; };

// A-matrix fragment (16x32 f16): lane<16 holds row m0+lane, K {0..7,16..23};
// lane>=16 holds row m0+lane-16, K {8..15,24..31}.  (ISA 7.12.2)
__device__ __forceinline__ v16h fragA(const _Float16* base, int stride, int m0,
                                      int k0, int lane) {
  const _Float16* p = base + (m0 + (lane & 15)) * stride + k0 + ((lane & 16) ? 8 : 0);
  FragU f;
  f.h[0] = *(const v8h*)(p);
  f.h[1] = *(const v8h*)(p + 16);
  return f.v;
}

// B-matrix fragment (32x16 f16), column-contiguous storage (B[k][n] with
// column n stored as a contiguous row): lane<16 holds col n0+lane, K {0..7,16..23};
// lane>=16 holds col n0+lane-16, K {8..15,24..31}.
__device__ __forceinline__ v16h fragB(const _Float16* base, int stride, int n0,
                                      int k0, int lane) {
  const _Float16* p = base + (n0 + (lane & 15)) * stride + k0 + ((lane & 16) ? 8 : 0);
  FragU f;
  f.h[0] = *(const v8h*)(p);
  f.h[1] = *(const v8h*)(p + 16);
  return f.v;
}

__device__ __forceinline__ v8f wmma16(v16h a, v16h b, v8f c) {
  return __builtin_amdgcn_wmma_f32_16x16x32_f16(false, a, false, b, (short)0, c,
                                                false, false);
}

// ---------------------------------------------------------------------------
// Kernel 1: exclusive prefix sum of agents_per_sample (2048 ints, one block).
// ---------------------------------------------------------------------------
__global__ __launch_bounds__(256) void scan_kernel(const int* __restrict__ agents,
                                                   int* __restrict__ offs) {
  __shared__ int part[256];
  const int t = threadIdx.x;
  const int base = t * 8;
  int vals[8];
  int tot = 0;
#pragma unroll
  for (int i = 0; i < 8; ++i) { vals[i] = agents[base + i]; tot += vals[i]; }
  part[t] = tot;
  __syncthreads();
  for (int off = 1; off < 256; off <<= 1) {
    int v = (t >= off) ? part[t - off] : 0;
    __syncthreads();
    part[t] += v;
    __syncthreads();
  }
  int run = part[t] - tot;  // exclusive chunk base
#pragma unroll
  for (int i = 0; i < 8; ++i) { offs[base + i] = run; run += vals[i]; }
}

// ---------------------------------------------------------------------------
// Kernel 2: convert w_in (384x128) and w_out (128x128) f32 -> f16 once.
// ---------------------------------------------------------------------------
__global__ __launch_bounds__(256) void cvt_kernel(const float* __restrict__ w_in,
                                                  const float* __restrict__ w_out,
                                                  _Float16* __restrict__ wi,
                                                  _Float16* __restrict__ wo) {
  int i = blockIdx.x * blockDim.x + threadIdx.x;
  const int NI = 384 * 128;
  const int NO = 128 * 128;
  if (i < NI) {
    wi[i] = (_Float16)w_in[i];
  } else if (i < NI + NO) {
    int j = i - NI;
    wo[j] = (_Float16)w_out[j];
  }
}

// ---------------------------------------------------------------------------
// Kernel 3: per-scene attention. One workgroup (4 waves) per scene.
// ---------------------------------------------------------------------------
__global__ __launch_bounds__(128) void mhsa_kernel(
    const float* __restrict__ att_in, const float* __restrict__ b_in,
    const float* __restrict__ b_out, const int* __restrict__ agents,
    const int* __restrict__ offs, const _Float16* __restrict__ w_in_h,
    const _Float16* __restrict__ w_out_h, float* __restrict__ out) {
  __shared__ _Float16 Xh[AMAX * XS];        // padded X (f16); reused as ctx
  __shared__ _Float16 Qh[AMAX * XS];        // Q  [agent][feat 0..127]
  __shared__ _Float16 Kh[AMAX * XS];        // K  [agent][feat 0..127]
  __shared__ _Float16 Vt[DIM * AS];         // V^T [feat][agent]
  __shared__ float    Sf[HEADS][16 * SFS];  // per-head 16-row f32 score scratch
  __shared__ _Float16 At[HEADS][AMAX * AS]; // per-head attention probs (f16)

  const int b = blockIdx.x;
  const int tid = threadIdx.x;
  const int lane = tid & 31;
  const int wv = tid >> 5;
  const int nA = agents[b];
  const int off = offs[b];
  const float scale = 0.17677669529663687f;  // 1/sqrt(32)

  // -------- Phase 0: gather ragged rows, zero-pad, convert to f16 ----------
#pragma unroll
  for (int i = 0; i < 16; ++i) {
    int slot = tid + i * 128;  // 2048 float4 slots = 64 rows x 32 float4
    int row = slot >> 5;
    int c4 = slot & 31;
    v4f val = {0.f, 0.f, 0.f, 0.f};
    if (row < nA)
      val = *(const v4f*)(att_in + (((size_t)(off + row)) << 7) + c4 * 4);
    v4h hv;
#pragma unroll
    for (int q = 0; q < 4; ++q) hv[q] = (_Float16)val[q];
    *(v4h*)(Xh + row * XS + c4 * 4) = hv;
  }
  __syncthreads();

  // -------- Phase 1: QKV = X @ w_in^T + b_in  (64x128 @ 128x384) -----------
  // Wave wv owns column tiles j = wv + 4t, t in 0..5 (24 tiles of 16 cols).
  {
    const int col = lane & 15;
    const int roff = (lane & 16) ? 8 : 0;
    for (int mi = 0; mi < 4; ++mi) {
      const int m0 = mi * 16;
      v16h a0 = fragA(Xh, XS, m0, 0, lane);
      v16h a1 = fragA(Xh, XS, m0, 32, lane);
      v16h a2 = fragA(Xh, XS, m0, 64, lane);
      v16h a3 = fragA(Xh, XS, m0, 96, lane);
      for (int t = 0; t < 6; ++t) {
        const int n0 = (wv + 4 * t) * 16;
        const float bias = b_in[n0 + col];
        v8f c = {bias, bias, bias, bias, bias, bias, bias, bias};
        c = wmma16(a0, fragB(w_in_h, 128, n0, 0, lane), c);
        c = wmma16(a1, fragB(w_in_h, 128, n0, 32, lane), c);
        c = wmma16(a2, fragB(w_in_h, 128, n0, 64, lane), c);
        c = wmma16(a3, fragB(w_in_h, 128, n0, 96, lane), c);
        if (n0 < 128) {  // Q
#pragma unroll
          for (int r = 0; r < 8; ++r)
            Qh[(m0 + r + roff) * XS + n0 + col] = (_Float16)c[r];
        } else if (n0 < 256) {  // K
#pragma unroll
          for (int r = 0; r < 8; ++r)
            Kh[(m0 + r + roff) * XS + (n0 - 128) + col] = (_Float16)c[r];
        } else {  // V stored transposed: Vt[feature][agent]
#pragma unroll
          for (int r = 0; r < 8; ++r)
            Vt[(n0 - 256 + col) * AS + (m0 + r + roff)] = (_Float16)c[r];
        }
      }
    }
  }
  __syncthreads();

  // -------- Phase 2: scores = Q K^T / sqrt(HD); masked softmax -------------
  // Wave wv == head h. S tile K-dim = HD = 32 -> single WMMA per 16x16 tile.
  {
    const int h = wv;
    const int kq = h * HD;
    float* sf = Sf[h];
    _Float16* at = At[h];
    const int col = lane & 15;
    const int roff = (lane & 16) ? 8 : 0;
    for (int mi = 0; mi < 4; ++mi) {
      const int m0 = mi * 16;
      v16h a = fragA(Qh, XS, m0, kq, lane);
#pragma unroll
      for (int ni = 0; ni < 4; ++ni) {
        v8f c = {0.f, 0.f, 0.f, 0.f, 0.f, 0.f, 0.f, 0.f};
        c = wmma16(a, fragB(Kh, XS, ni * 16, kq, lane), c);
#pragma unroll
        for (int r = 0; r < 8; ++r)
          sf[(r + roff) * SFS + ni * 16 + col] = c[r];
      }
      // Softmax over keys for these 16 rows (lanes 0..15, one row each).
      if (lane < 16) {
        const float* srow = sf + lane * SFS;
        float mx = -1e30f;
        for (int cc = 0; cc < AMAX; ++cc)
          if (cc < nA) mx = fmaxf(mx, srow[cc]);
        float sum = 0.f;
        _Float16* arow = at + (m0 + lane) * AS;
        for (int cc = 0; cc < AMAX; ++cc) {
          float e = (cc < nA) ? __expf((srow[cc] - mx) * scale) : 0.f;
          sum += e;
          arow[cc] = (_Float16)e;
        }
        const float inv = 1.f / sum;
        for (int cc = 0; cc < AMAX; ++cc)
          arow[cc] = (_Float16)((float)arow[cc] * inv);
      }
    }
  }

  // -------- Phase 3: ctx_h = attn_h @ V_h  (64x64 @ 64x32) -----------------
  // Reads At (own wave) and Vt (synced after phase 1); writes ctx into Xh
  // (X is dead for every wave after the phase-1 barrier).
  {
    const int h = wv;
    const _Float16* vtb = Vt + h * HD * AS;
    const _Float16* at = At[h];
    const int col = lane & 15;
    const int roff = (lane & 16) ? 8 : 0;
    for (int mi = 0; mi < 4; ++mi) {
      const int m0 = mi * 16;
      v16h a0 = fragA(at, AS, m0, 0, lane);
      v16h a1 = fragA(at, AS, m0, 32, lane);
#pragma unroll
      for (int di = 0; di < 2; ++di) {
        const int d0 = di * 16;
        v8f c = {0.f, 0.f, 0.f, 0.f, 0.f, 0.f, 0.f, 0.f};
        c = wmma16(a0, fragB(vtb, AS, d0, 0, lane), c);
        c = wmma16(a1, fragB(vtb, AS, d0, 32, lane), c);
#pragma unroll
        for (int r = 0; r < 8; ++r)
          Xh[(m0 + r + roff) * XS + h * HD + d0 + col] = (_Float16)c[r];
      }
    }
  }
  __syncthreads();

  // -------- Phase 4: out = ctx @ w_out^T + b_out, mask rows, store f32 -----
  {
    const int col = lane & 15;
    const int roff = (lane & 16) ? 8 : 0;
    for (int mi = 0; mi < 4; ++mi) {
      const int m0 = mi * 16;
      v16h a0 = fragA(Xh, XS, m0, 0, lane);
      v16h a1 = fragA(Xh, XS, m0, 32, lane);
      v16h a2 = fragA(Xh, XS, m0, 64, lane);
      v16h a3 = fragA(Xh, XS, m0, 96, lane);
#pragma unroll
      for (int t = 0; t < 2; ++t) {
        const int n0 = (wv + 4 * t) * 16;
        const float bias = b_out[n0 + col];
        v8f c = {bias, bias, bias, bias, bias, bias, bias, bias};
        c = wmma16(a0, fragB(w_out_h, 128, n0, 0, lane), c);
        c = wmma16(a1, fragB(w_out_h, 128, n0, 32, lane), c);
        c = wmma16(a2, fragB(w_out_h, 128, n0, 64, lane), c);
        c = wmma16(a3, fragB(w_out_h, 128, n0, 96, lane), c);
#pragma unroll
        for (int r = 0; r < 8; ++r) {
          const int m = m0 + r + roff;
          const float v = (m < nA) ? c[r] : 0.f;
          out[((size_t)b * AMAX + m) * DIM + n0 + col] = v;
        }
      }
    }
  }
}

// ---------------------------------------------------------------------------
extern "C" void kernel_launch(void* const* d_in, const int* in_sizes, int n_in,
                              void* d_out, int out_size, void* d_ws,
                              size_t ws_size, hipStream_t stream) {
  (void)in_sizes; (void)n_in; (void)out_size; (void)ws_size;
  const float* att_in = (const float*)d_in[0];
  const float* w_in   = (const float*)d_in[1];
  const float* b_in   = (const float*)d_in[2];
  const float* w_out  = (const float*)d_in[3];
  const float* b_out  = (const float*)d_in[4];
  const int*   agents = (const int*)d_in[5];
  float* outp = (float*)d_out;

  // Workspace layout: offsets (2048 int), w_in f16, w_out f16.
  int* offs = (int*)d_ws;
  _Float16* w_in_h  = (_Float16*)((char*)d_ws + 8192);
  _Float16* w_out_h = (_Float16*)((char*)d_ws + 8192 + 384 * 128 * 2);

  scan_kernel<<<1, 256, 0, stream>>>(agents, offs);
  cvt_kernel<<<256, 256, 0, stream>>>(w_in, w_out, w_in_h, w_out_h);
  mhsa_kernel<<<NSCENE, 128, 0, stream>>>(att_in, b_in, b_out, agents, offs,
                                          w_in_h, w_out_h, outp);
}
